// GraphModel_29197187678852
// MI455X (gfx1250) — compile-verified, matrix operand order checked by reading, not run on hardware
//
#include <hip/hip_runtime.h>
#include <math.h>

// ---------------------------------------------------------------------------
// GVP-GNN dependency parser forward pass for MI455X (gfx1250, wave32, WMMA).
// All dense contractions use v_wmma_f32_16x16x32_bf16 (f32 accumulate).
// B operands pre-swizzled to bf16 fragment order -> branch-free b128 loads.
// Zero-padding invariants make every GEMM epilogue branch-free.
// ---------------------------------------------------------------------------

#define NN 8192
#define EE 16384
#define BB 128

typedef __bf16 bf16;
typedef __attribute__((ext_vector_type(16))) __bf16 v16bf;
typedef __attribute__((ext_vector_type(2)))  __bf16 v2bf;
typedef __attribute__((ext_vector_type(8)))  float  v8f;

struct GvpW { const float *wh, *ws, *wsb, *wv, *wsv, *wsvb; };

// K index packing for 16-bit A/B fragments (ISA 7.12.2): element pair (2v,2v+1)
// covers K = base..base+1 with base = (v<4 ? 2v : 16+2(v-4)) + 8*(lane/16).
__device__ __forceinline__ int kmap16(int e, int half) {
  int v = e >> 1, p = e & 1;
  int base = (v < 4) ? (v * 2) : (16 + (v - 4) * 2);
  return base + half * 8 + p;
}

__device__ __forceinline__ v8f wmma_bf16(v16bf a, v16bf b, v8f c) {
  return __builtin_amdgcn_wmma_f32_16x16x32_bf16(false, a, false, b, (short)0, c,
                                                 false, false);
}

// A fragment from f32 memory (row-major, lda floats, rows = 16 tile rows).
__device__ __forceinline__ v16bf load_a_f32(const float* A, int lda, int k0, int lane) {
  int m = lane & 15, half = (lane >> 4) & 1;
  const float* row = A + m * lda + k0 + half * 8;
  v16bf f;
#pragma unroll
  for (int v = 0; v < 8; ++v) {
    int base = (v < 4) ? (v * 2) : (16 + (v - 4) * 2);
    float2 x = *(const float2*)(row + base);
    f[2 * v] = (bf16)x.x;
    f[2 * v + 1] = (bf16)x.y;
  }
  return f;
}

// Fragment from bf16 LDS tile (A rows = M, or B^T rows = N, identical math).
__device__ __forceinline__ v16bf load_frag_bf(const bf16* A, int lda, int k0, int lane) {
  int m = lane & 15, half = (lane >> 4) & 1;
  const bf16* row = A + m * lda + k0 + half * 8;
  v16bf f;
#pragma unroll
  for (int v = 0; v < 8; ++v) {
    int base = (v < 4) ? (v * 2) : (16 + (v - 4) * 2);
    v2bf x = *(const v2bf*)(row + base);
    f[2 * v] = x.x;
    f[2 * v + 1] = x.y;
  }
  return f;
}

// B fragment from pre-swizzled global bf16: one 32B contiguous chunk per lane.
__device__ __forceinline__ v16bf load_b_swz(const bf16* Bp, int NT, int kt, int nt, int lane) {
  return *(const v16bf*)(Bp + (((size_t)kt * NT + nt) << 9) + lane * 16);
}

__device__ __forceinline__ float sigm(float x) { return 1.f / (1.f + __expf(-x)); }

// ---------------------------------------------------------------------------
// Weight pre-swizzle: W[K x N] (f32, nmat matrices) -> fragment-ordered bf16,
// padded to KT*32 x NT*16 with zeros.
// ---------------------------------------------------------------------------
__global__ void k_prep_b(const float* W, bf16* out, int K, int N, int KT, int NT,
                         int nmat) {
  long long i = (long long)blockIdx.x * blockDim.x + threadIdx.x;
  long long per = (long long)KT * NT * 512;
  if (i >= per * nmat) return;
  int mat = (int)(i / per);
  long long r = i - (long long)mat * per;
  int tile = (int)(r >> 9);
  int lane = (int)((r >> 4) & 31);
  int e = (int)(r & 15);
  int kt = tile / NT, nt = tile - kt * NT;
  int k = kt * 32 + kmap16(e, (lane >> 4) & 1);
  int n = nt * 16 + (lane & 15);
  float x = (k < K && n < N) ? W[(long long)mat * K * N + (long long)k * N + n] : 0.f;
  out[i] = (bf16)x;
}

// ---------------------------------------------------------------------------
// Utility kernels
// ---------------------------------------------------------------------------
__global__ void k_zero(float* p, long long n) {
  long long i = (long long)blockIdx.x * blockDim.x + threadIdx.x;
  if (i < n) p[i] = 0.f;
}

__global__ void k_pad_copy(const float* src, float* dst, int rows, int cols, int stride) {
  int i = blockIdx.x * blockDim.x + threadIdx.x;
  if (i >= rows * stride) return;
  int r = i / stride, c = i - r * stride;
  dst[i] = (c < cols) ? src[r * cols + c] : 0.f;
}

__global__ void k_unpad(const float* s, float* out, int n) {
  int i = blockIdx.x * blockDim.x + threadIdx.x;
  if (i >= n * 100) return;
  int r = i / 100, c = i - r * 100;
  out[i] = s[r * 128 + c];
}

__global__ void k_count(const int* ei, float* cnt, int E) {
  int e = blockIdx.x * blockDim.x + threadIdx.x;
  if (e < E) atomicAdd(&cnt[ei[E + e]], 1.f);
}

// ---------------------------------------------------------------------------
// Fused per-edge GVP message chain m0->m1->m2, atomic scatter to (ds, dv).
// ---------------------------------------------------------------------------
#define CT 256
__device__ void gvp_edge_stage(float (*vm)[304], float (*vh)[304], float* sb,
                               float* sout, float* vn, float* gate, GvpW w,
                               int si, int vi, int h, int so, int vo,
                               int relu_s, int sig_gate, int tid) {
  for (int idx = tid; idx < h * 300; idx += CT) {
    int hh = idx / 300, d = idx - hh * 300;
    float acc = 0.f;
    for (int v = 0; v < vi; ++v) acc += vm[v][d] * w.wh[v * h + hh];
    vh[hh][d] = acc;
  }
  __syncthreads();
  {
    int g = tid >> 4, l = tid & 15;
    for (int hh = g; hh < h; hh += 16) {
      float acc = 0.f;
      for (int d = l; d < 300; d += 16) { float x = vh[hh][d]; acc += x * x; }
      acc += __shfl_xor(acc, 1, 16);
      acc += __shfl_xor(acc, 2, 16);
      acc += __shfl_xor(acc, 4, 16);
      acc += __shfl_xor(acc, 8, 16);
      if (l == 0) vn[hh] = sqrtf(fmaxf(acc, 1e-8f));
    }
  }
  __syncthreads();
  for (int o = tid; o < so; o += CT) {
    float acc = w.wsb[o];
    for (int i = 0; i < si; ++i) acc += sb[i] * w.ws[i * so + o];
    for (int hh = 0; hh < h; ++hh) acc += vn[hh] * w.ws[(si + hh) * so + o];
    sout[o] = acc;
  }
  __syncthreads();
  for (int o2 = tid; o2 < vo; o2 += CT) {
    float acc = w.wsvb[o2];
    for (int o = 0; o < so; ++o) {
      float sv = sout[o];
      if (sig_gate) sv = sigm(sv);
      acc += sv * w.wsv[o * vo + o2];
    }
    gate[o2] = sigm(acc);
  }
  __syncthreads();
  for (int idx = tid; idx < vo * 300; idx += CT) {
    int o2 = idx / 300, d = idx - o2 * 300;
    float acc = 0.f;
    for (int hh = 0; hh < h; ++hh) acc += vh[hh][d] * w.wv[hh * vo + o2];
    vm[o2][d] = acc * gate[o2];
  }
  for (int o = tid; o < so; o += CT) {
    float x = sout[o];
    sb[o] = relu_s ? fmaxf(x, 0.f) : x;
  }
  __syncthreads();
}

__global__ __launch_bounds__(256) void k_conv_edge(
    const float* sN, const float* vN, const int* ei, const float* es,
    const float* ev, GvpW m0, GvpW m1, GvpW m2, float* ds, float* dv, int E) {
  __shared__ float vm[17][304];
  __shared__ float vh[17][304];
  __shared__ float sb[256];
  __shared__ float sout[112];
  __shared__ float vn[20];
  __shared__ float gate[16];
  int e = blockIdx.x, tid = threadIdx.x;
  int src = ei[e], dst = ei[E + e];
  // gfx1250 prefetch (global_prefetch_b8) of the gathered v-channel rows
  {
    const char* ps = (const char*)(vN + (long long)src * 2400);
    const char* pd = (const char*)(vN + (long long)dst * 2400);
    __builtin_prefetch(ps + tid * 37, 0, 1);   // cover ~9.4KB row
    __builtin_prefetch(pd + tid * 37, 0, 1);
  }
  for (int idx = tid; idx < 17 * 300; idx += CT) {
    int r = idx / 300, d = idx - r * 300;
    float x;
    if (r < 8) x = vN[(long long)src * 2400 + r * 300 + d];
    else if (r == 8) x = ev[(long long)e * 300 + d];
    else x = vN[(long long)dst * 2400 + (r - 9) * 300 + d];
    vm[r][d] = x;
  }
  for (int i = tid; i < 232; i += CT) {
    float x;
    if (i < 100) x = sN[src * 128 + i];
    else if (i < 132) x = es[e * 32 + (i - 100)];
    else x = sN[dst * 128 + (i - 132)];
    sb[i] = x;
  }
  __syncthreads();
  gvp_edge_stage(vm, vh, sb, sout, vn, gate, m0, 232, 17, 17, 100, 8, 1, 1, tid);
  gvp_edge_stage(vm, vh, sb, sout, vn, gate, m1, 100, 8, 8, 100, 8, 1, 1, tid);
  gvp_edge_stage(vm, vh, sb, sout, vn, gate, m2, 100, 8, 8, 100, 8, 0, 0, tid);
  for (int i = tid; i < 100; i += CT) atomicAdd(&ds[dst * 128 + i], sb[i]);
  for (int idx = tid; idx < 2400; idx += CT)
    atomicAdd(&dv[(long long)dst * 2400 + idx], vm[idx / 300][idx % 300]);
}

// ---------------------------------------------------------------------------
// Residual + LayerNorm (s) + vector RMS norm (v); one wave per node.
// Branch-free s path: cols 100..127 of sN/ds are zero; g/b padded to 128 with
// zeros so the store rewrites zeros there. var corrected by -28*mu^2.
// ---------------------------------------------------------------------------
__global__ __launch_bounds__(256) void k_residual_ln(
    float* sN, float* vN, const float* ds, const float* dv, const float* cnt,
    const float* g, const float* b, int N) {
  int wv_ = threadIdx.x >> 5, lane = threadIdx.x & 31;
  int n = blockIdx.x * 8 + wv_;
  float inv = cnt ? 1.f / fmaxf(cnt[n], 1.f) : 1.f;
  float local[4];
  float sum = 0.f;
#pragma unroll
  for (int t = 0; t < 4; ++t) {
    int i = lane + t * 32;
    float x = sN[n * 128 + i] + ds[n * 128 + i] * inv;
    local[t] = x;
    sum += x;
  }
  for (int m = 16; m; m >>= 1) sum += __shfl_xor(sum, m, 32);
  float mu = sum / 100.f;
  float var = 0.f;
#pragma unroll
  for (int t = 0; t < 4; ++t) { float d = local[t] - mu; var += d * d; }
  for (int m = 16; m; m >>= 1) var += __shfl_xor(var, m, 32);
  var -= 28.f * mu * mu;  // remove contribution of the 28 zero pad columns
  float rstd = rsqrtf(var / 100.f + 1e-5f);
#pragma unroll
  for (int t = 0; t < 4; ++t) {
    int i = lane + t * 32;
    sN[n * 128 + i] = (local[t] - mu) * rstd * g[i] + b[i];
  }
  float vsum = 0.f;
  for (int r = 0; r < 8; ++r) {
    float ss = 0.f;
    for (int d = lane; d < 300; d += 32) {
      long long o = (long long)n * 2400 + r * 300 + d;
      float x = vN[o] + dv[o] * inv;
      ss += x * x;
    }
    for (int m = 16; m; m >>= 1) ss += __shfl_xor(ss, m, 32);
    vsum += fmaxf(ss, 1e-8f);
  }
  float rv = rsqrtf(vsum / 8.f);
  for (int idx = lane; idx < 2400; idx += 32) {
    long long o = (long long)n * 2400 + idx;
    vN[o] = (vN[o] + dv[o] * inv) * rv;
  }
}

// ---------------------------------------------------------------------------
// s-only GVP head (vi=8,h=8,K=108 padded 128): WMMA over 16-node tiles.
// Zero-padded weights+bias make cols >= so come out exactly 0 -> no guards.
// SACT: 0 none, 1 relu, 2 tanh (compile-time -> straight-line epilogue)
// ---------------------------------------------------------------------------
template <int SACT>
__global__ __launch_bounds__(256) void k_gvp_head(
    const float* sN, const float* vN, const float* wh, const bf16* wsp,
    const float* pbias, float* out, int ostride, int NT) {
  __shared__ float X[16][128];
  int tid = threadIdx.x, n0 = blockIdx.x * 16;
  __builtin_prefetch((const char*)(vN + (long long)n0 * 2400) + tid * 600, 0, 1);
  // stage s row (cols >= 100 are zero in s_cur)
  for (int i = tid; i < 16 * 128; i += 256) {
    int nl = i >> 7, c = i & 127;
    X[nl][c] = sN[(n0 + nl) * 128 + c];
  }
  __syncthreads();
  // vn overwrite cols 100..107
  if (tid < 128) {
    int nl = tid >> 3, hh = tid & 7;
    const float* vp = vN + (long long)(n0 + nl) * 2400;
    float acc = 0.f;
    for (int d = 0; d < 300; ++d) {
      float t = 0.f;
#pragma unroll
      for (int v = 0; v < 8; ++v) t += vp[v * 300 + d] * wh[v * 8 + hh];
      acc += t * t;
    }
    X[nl][100 + hh] = sqrtf(fmaxf(acc, 1e-8f));
  }
  __syncthreads();
  int lane = tid & 31, wv_ = tid >> 5;
  for (int nt = wv_; nt < NT; nt += 8) {
    v8f c = {};
#pragma unroll
    for (int kt = 0; kt < 4; ++kt) {
      v16bf a = load_a_f32(&X[0][0], 128, kt * 32, lane);
      v16bf bb = load_b_swz(wsp, NT, kt, nt, lane);
      c = wmma_bf16(a, bb, c);
    }
    int col = nt * 16 + (lane & 15), half = (lane >> 4) & 1;
    float bias = pbias[col];
#pragma unroll
    for (int r = 0; r < 8; ++r) {
      int nl = r + half * 8;
      float x = c[r] + bias;
      if (SACT == 1) x = fmaxf(x, 0.f);
      if (SACT == 2) x = tanhf(x);
      out[(long long)(n0 + nl) * ostride + col] = x;
    }
  }
}

// ---------------------------------------------------------------------------
// Fused FF block (ff0 -> ff1): vector paths collapsed into per-node 8x8
// combined matrices; WMMA for both scalar GEMMs (pre-swizzled B).
// ---------------------------------------------------------------------------
__global__ __launch_bounds__(256) void k_ff_fused(
    const float* sN, const float* vN, GvpW f0, GvpW f1, const bf16* ws0p,
    const bf16* ws1p, const float* f1pb, float* ds, float* dv) {
  __shared__ float FS[16][416];
  __shared__ float X0[16][128];
  __shared__ float S1[16][112];
  __shared__ float g0[16][16];
  __shared__ float g1[16][8];
  __shared__ float C0s[8][16];
  __shared__ float C1s[16][8];
  __shared__ float Gs[16][8][16];
  __shared__ float Mns[16][8][8];
  int tid = threadIdx.x, n0 = blockIdx.x * 16;
  int lane = tid & 31, wv_ = tid >> 5;
  __builtin_prefetch((const char*)(vN + (long long)n0 * 2400) + tid * 600, 0, 1);
  // stage s rows (cols >= 100 zero)
  for (int i = tid; i < 16 * 128; i += 256) {
    int nl = i >> 7, c = i & 127;
    X0[nl][c] = sN[(n0 + nl) * 128 + c];
  }
  __syncthreads();
  // vn0 (h=16) overwrite cols 100..115
  {
    int nl = tid >> 4, hh = tid & 15;
    const float* vp = vN + (long long)(n0 + nl) * 2400;
    float acc = 0.f;
    for (int d = 0; d < 300; ++d) {
      float t = 0.f;
#pragma unroll
      for (int v = 0; v < 8; ++v) t += vp[v * 300 + d] * f0.wh[v * 16 + hh];
      acc += t * t;
    }
    X0[nl][100 + hh] = sqrtf(fmaxf(acc, 1e-8f));
  }
  if (tid < 128) {  // C0 = wh0 @ wv0  [8x16]
    int v = tid >> 4, o = tid & 15;
    float a = 0.f;
    for (int hh = 0; hh < 16; ++hh) a += f0.wh[v * 16 + hh] * f0.wv[hh * 16 + o];
    C0s[v][o] = a;
  } else {          // C1 = wh1 @ wv1  [16x8]
    int t = tid - 128, v = t >> 3, o = t & 7;
    float a = 0.f;
    for (int hh = 0; hh < 16; ++hh) a += f1.wh[v * 16 + hh] * f1.wv[hh * 8 + o];
    C1s[v][o] = a;
  }
  __syncthreads();
  // GEMM1: FS = X0 @ ws0 + b0  (raw, 16x400), NT=25
  for (int nt = wv_; nt < 25; nt += 8) {
    v8f c = {};
#pragma unroll
    for (int kt = 0; kt < 4; ++kt) {
      v16bf a = load_a_f32(&X0[0][0], 128, kt * 32, lane);
      v16bf bb = load_b_swz(ws0p, 25, kt, nt, lane);
      c = wmma_bf16(a, bb, c);
    }
    int col = nt * 16 + (lane & 15), half = (lane >> 4) & 1;
    float bias = f0.wsb[col];
#pragma unroll
    for (int r = 0; r < 8; ++r) FS[r + half * 8][col] = c[r] + bias;
  }
  __syncthreads();
  {
    int nl = tid >> 4, o2 = tid & 15;
    float acc = f0.wsvb[o2];
    for (int o = 0; o < 400; ++o) acc += sigm(FS[nl][o]) * f0.wsv[o * 16 + o2];
    g0[nl][o2] = sigm(acc);
  }
  __syncthreads();
  for (int i = tid; i < 16 * 400; i += 256) {
    int nl = i / 400, c = i - nl * 400;
    FS[nl][c] = fmaxf(FS[nl][c], 0.f);
  }
  for (int i = tid; i < 16 * 8 * 16; i += 256) {
    int nl = i >> 7, rem = i & 127, v = rem >> 4, h1 = rem & 15;
    float a = 0.f;
    for (int vo0 = 0; vo0 < 16; ++vo0)
      a += C0s[v][vo0] * g0[nl][vo0] * f1.wh[vo0 * 16 + h1];
    Gs[nl][v][h1] = a;
  }
  __syncthreads();
  {
    int nl = tid >> 4, h1 = tid & 15;
    const float* vp = vN + (long long)(n0 + nl) * 2400;
    float acc = 0.f;
    for (int d = 0; d < 300; ++d) {
      float t = 0.f;
#pragma unroll
      for (int v = 0; v < 8; ++v) t += vp[v * 300 + d] * Gs[nl][v][h1];
      acc += t * t;
    }
    FS[nl][400 + h1] = sqrtf(fmaxf(acc, 1e-8f));
  }
  __syncthreads();
  // GEMM2: S1 = FS(416) @ ws1 + b1 (raw; cols 100..111 come out 0)
  for (int nt = wv_; nt < 7; nt += 8) {
    v8f c = {};
#pragma unroll
    for (int kt = 0; kt < 13; ++kt) {
      v16bf a = load_a_f32(&FS[0][0], 416, kt * 32, lane);
      v16bf bb = load_b_swz(ws1p, 8, kt, nt, lane);
      c = wmma_bf16(a, bb, c);
    }
    int col = nt * 16 + (lane & 15), half = (lane >> 4) & 1;
    float bias = f1pb[col];
#pragma unroll
    for (int r = 0; r < 8; ++r) {
      int nl = r + half * 8;
      float x = c[r] + bias;
      S1[nl][col] = x;
      ds[(long long)(n0 + nl) * 128 + col] = x;
    }
  }
  __syncthreads();
  if (tid < 128) {
    int nl = tid >> 3, o2 = tid & 7;
    float acc = f1.wsvb[o2];
    for (int o = 0; o < 100; ++o) acc += S1[nl][o] * f1.wsv[o * 8 + o2];
    g1[nl][o2] = sigm(acc);
  }
  for (int i = tid; i < 16 * 64; i += 256) {
    int nl = i >> 6, rem = i & 63, v = rem >> 3, o = rem & 7;
    float a = 0.f;
    for (int vo0 = 0; vo0 < 16; ++vo0)
      a += C0s[v][vo0] * g0[nl][vo0] * C1s[vo0][o];
    Mns[nl][v][o] = a;
  }
  __syncthreads();
  for (int i = tid; i < 16 * 2400; i += 256) {
    int nl = i / 2400, rem = i - nl * 2400, o = rem / 300, d = rem - o * 300;
    const float* vp = vN + (long long)(n0 + nl) * 2400;
    float a = 0.f;
#pragma unroll
    for (int v = 0; v < 8; ++v) a += vp[v * 300 + d] * Mns[nl][v][o];
    dv[(long long)(n0 + nl) * 2400 + rem] = a * g1[nl][o];
  }
}

// ---------------------------------------------------------------------------
// C[M,128] = A[M,128 f32] @ Bswz (K=128, NT=8). Padded B -> cols>=Ncols are 0.
// ---------------------------------------------------------------------------
__global__ __launch_bounds__(32) void k_gemm_swz(
    const float* A, const bf16* Bp, float* C) {
  int lane = threadIdx.x;
  int m0 = blockIdx.x * 16, nt = blockIdx.y;
  v8f c = {};
#pragma unroll
  for (int kt = 0; kt < 4; ++kt) {
    v16bf a = load_a_f32(A + (long long)m0 * 128, 128, kt * 32, lane);
    v16bf bb = load_b_swz(Bp, 8, kt, nt, lane);
    c = wmma_bf16(a, bb, c);
  }
  int col = nt * 16 + (lane & 15), half = (lane >> 4) & 1;
#pragma unroll
  for (int r = 0; r < 8; ++r) {
    int row = m0 + r + half * 8;
    C[(long long)row * 128 + col] = c[r];
  }
}

// ---------------------------------------------------------------------------
// Per-batch biaffine arc scores + row softmax: s_arc[b,i,j].
// ---------------------------------------------------------------------------
__global__ __launch_bounds__(256) void k_arc_softmax(
    const float* ha, const float* ad2, float* s_arc) {
  __shared__ bf16 HA[64][128];
  __shared__ bf16 AD[64][128];
  __shared__ float LG[64][64];
  int b = blockIdx.x, tid = threadIdx.x;
  for (int i = tid; i < 64 * 128; i += 256) {
    int r = i >> 7, c = i & 127;
    long long n = (long long)(b * 64 + r) * 128 + c;
    HA[r][c] = (bf16)ha[n];
    AD[r][c] = (bf16)ad2[n];
  }
  __syncthreads();
  int lane = tid & 31, wv_ = tid >> 5;
  for (int t = wv_; t < 16; t += 8) {
    int mt = t >> 2, nt = t & 3;
    v8f c = {};
#pragma unroll
    for (int kt = 0; kt < 4; ++kt) {
      v16bf a = load_frag_bf(&HA[mt * 16][0], 128, kt * 32, lane);
      v16bf bb = load_frag_bf(&AD[nt * 16][0], 128, kt * 32, lane);
      c = wmma_bf16(a, bb, c);
    }
    int col = nt * 16 + (lane & 15), half = (lane >> 4) & 1;
#pragma unroll
    for (int r = 0; r < 8; ++r) LG[mt * 16 + r + half * 8][col] = c[r];
  }
  __syncthreads();
  for (int r = wv_; r < 64; r += 8) {
    float x0 = LG[r][lane], x1 = LG[r][lane + 32];
    float mx = fmaxf(x0, x1);
    for (int m = 16; m; m >>= 1) mx = fmaxf(mx, __shfl_xor(mx, m, 32));
    float e0 = __expf(x0 - mx), e1 = __expf(x1 - mx);
    float s = e0 + e1;
    for (int m = 16; m; m >>= 1) s += __shfl_xor(s, m, 32);
    float inv = 1.f / s;
    s_arc[((long long)b * 64 + r) * 64 + lane] = e0 * inv;
    s_arc[((long long)b * 64 + r) * 64 + lane + 32] = e1 * inv;
  }
}

// ---------------------------------------------------------------------------
// Fused dep_k + s_lab per (b,k): DK = lab_dep[b] @ lab_w[k]; logits[l,m] =
// DK[l,:] . h_lab[b,m,:]. Raw logits -> d_out scores region.
// ---------------------------------------------------------------------------
__global__ __launch_bounds__(256) void k_depk_slab(
    const float* ldp, const float* hl, const bf16* labwp, float* logits) {
  __shared__ bf16 LD[64][128];
  __shared__ bf16 HL[64][128];
  __shared__ bf16 DK[64][128];
  int b = blockIdx.x / 40, k = blockIdx.x % 40;
  int tid = threadIdx.x, lane = tid & 31, wv_ = tid >> 5;
  for (int i = tid; i < 64 * 128; i += 256) {
    int r = i >> 7, c = i & 127;
    long long n = (long long)(b * 64 + r) * 128 + c;
    LD[r][c] = (bf16)ldp[n];
    HL[r][c] = (bf16)hl[n];
  }
  __syncthreads();
  const bf16* Wk = labwp + (size_t)k * (4 * 8 * 512);
  for (int t = wv_; t < 32; t += 8) {   // 4 m-tiles x 8 n-tiles (Wk padded)
    int mt = t >> 3, nt = t & 7;
    v8f c = {};
#pragma unroll
    for (int kt = 0; kt < 4; ++kt) {
      v16bf a = load_frag_bf(&LD[mt * 16][0], 128, kt * 32, lane);
      v16bf bb = load_b_swz(Wk, 8, kt, nt, lane);
      c = wmma_bf16(a, bb, c);
    }
    int col = nt * 16 + (lane & 15), half = (lane >> 4) & 1;
#pragma unroll
    for (int r = 0; r < 8; ++r)
      DK[mt * 16 + r + half * 8][col] = (bf16)c[r];
  }
  __syncthreads();
  for (int t = wv_; t < 16; t += 8) {
    int mt = t >> 2, nt = t & 3;
    v8f c = {};
#pragma unroll
    for (int kt = 0; kt < 4; ++kt) {
      v16bf a = load_frag_bf(&DK[mt * 16][0], 128, kt * 32, lane);
      v16bf bb = load_frag_bf(&HL[nt * 16][0], 128, kt * 32, lane);
      c = wmma_bf16(a, bb, c);
    }
    int m_ = nt * 16 + (lane & 15), half = (lane >> 4) & 1;
#pragma unroll
    for (int r = 0; r < 8; ++r) {
      int l_ = mt * 16 + r + half * 8;
      logits[(((long long)b * 64 + l_) * 64 + m_) * 40 + k] = c[r];
    }
  }
}

// softmax over k (in place on logits) * s_arc
__global__ void k_scores_final(float* scores, const float* s_arc, int total) {
  int i = blockIdx.x * blockDim.x + threadIdx.x;
  if (i >= total) return;
  float* row = scores + (long long)i * 40;
  float mx = -1e30f;
  for (int k = 0; k < 40; ++k) mx = fmaxf(mx, row[k]);
  float e_[40], s = 0.f;
  for (int k = 0; k < 40; ++k) { e_[k] = __expf(row[k] - mx); s += e_[k]; }
  float a = s_arc[i] / s;
  for (int k = 0; k < 40; ++k) row[k] = e_[k] * a;
}

__global__ void k_pos_cls(const float* pos16, const float* W, const float* bia,
                          float* out, int N) {
  int i = blockIdx.x * blockDim.x + threadIdx.x;
  if (i >= N * 17) return;
  int n = i / 17, j = i - n * 17;
  float acc = bia[j];
  for (int c = 0; c < 16; ++c) acc += pos16[n * 16 + c] * W[c * 17 + j];
  out[i] = acc;
}

// ---------------------------------------------------------------------------
// Host launch
// ---------------------------------------------------------------------------
static inline GvpW gw(void* const* d_in, int i) {
  GvpW w;
  w.wh = (const float*)d_in[i];     w.ws = (const float*)d_in[i + 1];
  w.wsb = (const float*)d_in[i + 2]; w.wv = (const float*)d_in[i + 3];
  w.wsv = (const float*)d_in[i + 4]; w.wsvb = (const float*)d_in[i + 5];
  return w;
}

extern "C" void kernel_launch(void* const* d_in, const int* in_sizes, int n_in,
                              void* d_out, int out_size, void* d_ws, size_t ws_size,
                              hipStream_t stream) {
  (void)in_sizes; (void)n_in; (void)out_size; (void)ws_size;
  const int N = NN, E = EE, B = BB;
  const float* node_s = (const float*)d_in[0];
  const float* node_v = (const float*)d_in[1];
  const int*   ei     = (const int*)d_in[2];
  const float* edge_s = (const float*)d_in[3];
  const float* edge_v = (const float*)d_in[4];
  // params in setup_inputs() insertion order:
  // W1: m0@5 m1@11 m2@17 ff0@23 ff1@29 ln0(g,b)@35 ln1(g,b)@37
  // W2: base 39. pos@73. arc_head@76 arc_dep@79 lab_head@82 lab_dep@85.
  // pos_cls_w@88 pos_cls_b@89 arc_w@90 lab_w@91
  GvpW M0[2] = {gw(d_in, 5), gw(d_in, 39)};
  GvpW M1[2] = {gw(d_in, 11), gw(d_in, 45)};
  GvpW M2[2] = {gw(d_in, 17), gw(d_in, 51)};
  GvpW F0[2] = {gw(d_in, 23), gw(d_in, 57)};
  GvpW F1[2] = {gw(d_in, 29), gw(d_in, 63)};
  const float* LN0G[2] = {(const float*)d_in[35], (const float*)d_in[69]};
  const float* LN0B[2] = {(const float*)d_in[36], (const float*)d_in[70]};
  const float* LN1G[2] = {(const float*)d_in[37], (const float*)d_in[71]};
  const float* LN1B[2] = {(const float*)d_in[38], (const float*)d_in[72]};
  const float* pos_wh = (const float*)d_in[73];
  const float* pos_ws = (const float*)d_in[74];
  const float* pos_wsb = (const float*)d_in[75];
  const float* ah_wh = (const float*)d_in[76], *ah_ws = (const float*)d_in[77], *ah_wsb = (const float*)d_in[78];
  const float* ad_wh = (const float*)d_in[79], *ad_ws = (const float*)d_in[80], *ad_wsb = (const float*)d_in[81];
  const float* lh_wh = (const float*)d_in[82], *lh_ws = (const float*)d_in[83], *lh_wsb = (const float*)d_in[84];
  const float* ld_wh = (const float*)d_in[85], *ld_ws = (const float*)d_in[86], *ld_wsb = (const float*)d_in[87];
  const float* pcw = (const float*)d_in[88];
  const float* pcb = (const float*)d_in[89];
  const float* arc_w = (const float*)d_in[90];
  const float* lab_w = (const float*)d_in[91];

  float* Wf = (float*)d_ws;
  size_t off = 0;
  float* s_cur = Wf + off; off += (size_t)N * 128;
  float* v_cur = Wf + off; off += (size_t)N * 2400;
  float* ds    = Wf + off; off += (size_t)N * 128;
  float* dv    = Wf + off; off += (size_t)N * 2400;
  float* cnt   = Wf + off; off += (size_t)N;
  float* ha    = Wf + off; off += (size_t)N * 128;
  float* adp   = Wf + off; off += (size_t)N * 128;
  float* adp2  = Wf + off; off += (size_t)N * 128;
  float* hl    = Wf + off; off += (size_t)N * 128;
  float* ldp   = Wf + off; off += (size_t)N * 128;
  float* pos16 = Wf + off; off += (size_t)N * 16;
  float* sarc  = Wf + off; off += (size_t)B * 64 * 64;
  // padded biases / LN params (stride 128, zero-filled)
  float* pb_ah = Wf + off; off += 128;
  float* pb_ad = Wf + off; off += 128;
  float* pb_lh = Wf + off; off += 128;
  float* pb_ld = Wf + off; off += 128;
  float* pb_f1[2]; pb_f1[0] = Wf + off; off += 128;
                   pb_f1[1] = Wf + off; off += 128;
  float* pLN0G[2]; pLN0G[0] = Wf + off; off += 128; pLN0G[1] = Wf + off; off += 128;
  float* pLN0B[2]; pLN0B[0] = Wf + off; off += 128; pLN0B[1] = Wf + off; off += 128;
  float* pLN1G[2]; pLN1G[0] = Wf + off; off += 128; pLN1G[1] = Wf + off; off += 128;
  float* pLN1B[2]; pLN1B[0] = Wf + off; off += 128; pLN1B[1] = Wf + off; off += 128;
  off = (off + 7) & ~(size_t)7;  // 32B align for bf16 fragment loads
  bf16* Bw = (bf16*)(Wf + off);
  size_t boff = 0;
  bf16* posp = Bw + boff; boff += 4 * 1 * 512;
  bf16* ahp  = Bw + boff; boff += 4 * 8 * 512;
  bf16* adpp = Bw + boff; boff += 4 * 8 * 512;
  bf16* lhp  = Bw + boff; boff += 4 * 8 * 512;
  bf16* ldpp = Bw + boff; boff += 4 * 8 * 512;
  bf16* f0p[2]; f0p[0] = Bw + boff; boff += 4 * 25 * 512;
                f0p[1] = Bw + boff; boff += 4 * 25 * 512;
  bf16* f1p[2]; f1p[0] = Bw + boff; boff += 13 * 8 * 512;
                f1p[1] = Bw + boff; boff += 13 * 8 * 512;
  bf16* arcp = Bw + boff; boff += 4 * 8 * 512;
  bf16* labp = Bw + boff; boff += (size_t)40 * 4 * 8 * 512;

  float* out    = (float*)d_out;
  float* out_s  = out;                 // N*100
  float* out_v  = out + 819200;        // N*8*300
  float* out_p  = out + 20480000;      // N*17
  float* out_sc = out + 20619264;      // B*64*64*40

  // weight pre-swizzle (fragment-ordered bf16)
  k_prep_b<<<(4 * 1 * 512 + 255) / 256, 256, 0, stream>>>(pos_ws, posp, 108, 16, 4, 1, 1);
  k_prep_b<<<(4 * 8 * 512 + 255) / 256, 256, 0, stream>>>(ah_ws, ahp, 108, 100, 4, 8, 1);
  k_prep_b<<<(4 * 8 * 512 + 255) / 256, 256, 0, stream>>>(ad_ws, adpp, 108, 100, 4, 8, 1);
  k_prep_b<<<(4 * 8 * 512 + 255) / 256, 256, 0, stream>>>(lh_ws, lhp, 108, 100, 4, 8, 1);
  k_prep_b<<<(4 * 8 * 512 + 255) / 256, 256, 0, stream>>>(ld_ws, ldpp, 108, 100, 4, 8, 1);
  k_prep_b<<<(4 * 25 * 512 + 255) / 256, 256, 0, stream>>>(F0[0].ws, f0p[0], 116, 400, 4, 25, 1);
  k_prep_b<<<(4 * 25 * 512 + 255) / 256, 256, 0, stream>>>(F0[1].ws, f0p[1], 116, 400, 4, 25, 1);
  k_prep_b<<<(13 * 8 * 512 + 255) / 256, 256, 0, stream>>>(F1[0].ws, f1p[0], 416, 100, 13, 8, 1);
  k_prep_b<<<(13 * 8 * 512 + 255) / 256, 256, 0, stream>>>(F1[1].ws, f1p[1], 416, 100, 13, 8, 1);
  k_prep_b<<<(4 * 8 * 512 + 255) / 256, 256, 0, stream>>>(arc_w, arcp, 100, 100, 4, 8, 1);
  k_prep_b<<<(40 * 4 * 8 * 512 + 255) / 256, 256, 0, stream>>>(lab_w, labp, 100, 100, 4, 8, 40);

  // padded bias / LN staging
  k_pad_copy<<<1, 128, 0, stream>>>(ah_wsb, pb_ah, 1, 100, 128);
  k_pad_copy<<<1, 128, 0, stream>>>(ad_wsb, pb_ad, 1, 100, 128);
  k_pad_copy<<<1, 128, 0, stream>>>(lh_wsb, pb_lh, 1, 100, 128);
  k_pad_copy<<<1, 128, 0, stream>>>(ld_wsb, pb_ld, 1, 100, 128);
  for (int L = 0; L < 2; ++L) {
    k_pad_copy<<<1, 128, 0, stream>>>(F1[L].wsb, pb_f1[L], 1, 100, 128);
    k_pad_copy<<<1, 128, 0, stream>>>(LN0G[L], pLN0G[L], 1, 100, 128);
    k_pad_copy<<<1, 128, 0, stream>>>(LN0B[L], pLN0B[L], 1, 100, 128);
    k_pad_copy<<<1, 128, 0, stream>>>(LN1G[L], pLN1G[L], 1, 100, 128);
    k_pad_copy<<<1, 128, 0, stream>>>(LN1B[L], pLN1B[L], 1, 100, 128);
  }

  k_pad_copy<<<(N * 128 + 255) / 256, 256, 0, stream>>>(node_s, s_cur, N, 100, 128);
  hipMemcpyAsync(v_cur, node_v, (size_t)N * 2400 * sizeof(float),
                 hipMemcpyDeviceToDevice, stream);
  k_zero<<<(N + 255) / 256, 256, 0, stream>>>(cnt, N);
  k_count<<<(E + 255) / 256, 256, 0, stream>>>(ei, cnt, E);

  for (int L = 0; L < 2; ++L) {
    k_zero<<<(N * 128 + 255) / 256, 256, 0, stream>>>(ds, (long long)N * 128);
    k_zero<<<((long long)N * 2400 + 255) / 256, 256, 0, stream>>>(dv, (long long)N * 2400);
    k_conv_edge<<<E, 256, 0, stream>>>(s_cur, v_cur, ei, edge_s, edge_v,
                                       M0[L], M1[L], M2[L], ds, dv, E);
    k_residual_ln<<<N / 8, 256, 0, stream>>>(s_cur, v_cur, ds, dv, cnt,
                                             pLN0G[L], pLN0B[L], N);
    k_ff_fused<<<N / 16, 256, 0, stream>>>(s_cur, v_cur, F0[L], F1[L],
                                           f0p[L], f1p[L], pb_f1[L], ds, dv);
    k_residual_ln<<<N / 8, 256, 0, stream>>>(s_cur, v_cur, ds, dv, nullptr,
                                             pLN1G[L], pLN1B[L], N);
  }

  k_gvp_head<2><<<N / 16, 256, 0, stream>>>(s_cur, v_cur, pos_wh, posp, pos_wsb, pos16, 16, 1);
  k_gvp_head<1><<<N / 16, 256, 0, stream>>>(s_cur, v_cur, ah_wh, ahp, pb_ah, ha, 128, 8);
  k_gvp_head<1><<<N / 16, 256, 0, stream>>>(s_cur, v_cur, ad_wh, adpp, pb_ad, adp, 128, 8);
  k_gvp_head<1><<<N / 16, 256, 0, stream>>>(s_cur, v_cur, lh_wh, lhp, pb_lh, hl, 128, 8);
  k_gvp_head<1><<<N / 16, 256, 0, stream>>>(s_cur, v_cur, ld_wh, ldpp, pb_ld, ldp, 128, 8);

  k_gemm_swz<<<dim3(N / 16, 8), 32, 0, stream>>>(adp, arcp, adp2);
  k_arc_softmax<<<B, 256, 0, stream>>>(ha, adp2, sarc);
  k_depk_slab<<<B * 40, 256, 0, stream>>>(ldp, hl, labp, out_sc);
  k_scores_final<<<(B * 4096 + 255) / 256, 256, 0, stream>>>(out_sc, sarc, B * 4096);
  k_pos_cls<<<(N * 17 + 255) / 256, 256, 0, stream>>>(pos16, pcw, pcb, out_p, N);
  k_unpad<<<(N * 100 + 255) / 256, 256, 0, stream>>>(s_cur, out_s, N);
  hipMemcpyAsync(out_v, v_cur, (size_t)N * 2400 * sizeof(float),
                 hipMemcpyDeviceToDevice, stream);
}
